// SparseDIST_58823872086161
// MI455X (gfx1250) — matrix operand order repaired and checked by reading, not compiled, not run on hardware
//
#include <hip/hip_runtime.h>
#include <hip/hip_bf16.h>
#include <math.h>

// ---------------------------------------------------------------------------
// Problem constants (match reference)
// ---------------------------------------------------------------------------
#define NPTS    8192
#define KNN     12
#define CDIM    128
#define LLAYERS 6
#define FFDIM   512
#define MEDGE   (NPTS * KNN)      // 98304 edge rows, divisible by 128

typedef __attribute__((ext_vector_type(16))) _Float16 v16h;
typedef __attribute__((ext_vector_type(8)))  _Float16 v8h;
typedef __attribute__((ext_vector_type(4)))  _Float16 v4h;
typedef __attribute__((ext_vector_type(8)))  float    v8f;

// ---------------------------------------------------------------------------
// WMMA fragment loaders (CDNA5 16x16x32 f16, wave32 layouts per ISA 7.12.2)
// A (16x32, row-major src): lane L -> row = L&15, h=L>>4;
//   halves 0..7  = K: k0+8h .. k0+8h+7      (contiguous 16B)
//   halves 8..15 = K: k0+16+8h .. +7        (contiguous 16B)
// B (32x16) from N-major weight copy Wt[fout][fin]: lane L -> col = L&15,
//   halves 0..7 = K: k0+8h .. +7, halves 8..15 = K: k0+16+8h .. +7 (same as A)
// ---------------------------------------------------------------------------
__device__ __forceinline__ v16h load_a_frag(const _Float16* __restrict__ A,
                                            int lda, int row0, int k0, int lane) {
    const int m = row0 + (lane & 15);
    const int h = lane >> 4;
    const _Float16* p = A + (size_t)m * lda + k0 + h * 8;
    v8h lo = *(const v8h*)p;
    v8h hi = *(const v8h*)(p + 16);
    v16h f;
#pragma unroll
    for (int i = 0; i < 8; ++i) { f[i] = lo[i]; f[i + 8] = hi[i]; }
    return f;
}

__device__ __forceinline__ v16h load_b_frag(const _Float16* __restrict__ Wt,
                                            int ldw, int col0, int k0, int lane) {
    const int n = col0 + (lane & 15);
    const int h = lane >> 4;
    const _Float16* p = Wt + (size_t)n * ldw + k0 + h * 8;
    v8h lo = *(const v8h*)p;
    v8h hi = *(const v8h*)(p + 16);
    v16h f;
#pragma unroll
    for (int i = 0; i < 8; ++i) { f[i] = lo[i]; f[i + 8] = hi[i]; }
    return f;
}

// ---------------------------------------------------------------------------
// Generic WMMA GEMM: out[M,Nout] = A[M,Kin](f16) @ Wt^T + bias, 256 thr = 8 waves,
// block tile 128 rows x 128 cols (blockIdx.y = col block). Wave: 16 rows x 128 cols.
// MODE 0: outF = y
// MODE 1: outH = (f16) gelu_exact(y)
// MODE 2: outF = sigmoid(acc2 + bias2) * y           (dual accumulators)
// MODE 3: outF += sigmoid(y) * aux                   (gate applied to buffer)
// MODE 4: outF += y                                  (residual accumulate)
// ---------------------------------------------------------------------------
template <int MODE>
__global__ __launch_bounds__(256) void gemm128(
    const _Float16* __restrict__ A, int Kin,
    const _Float16* __restrict__ Wt, const float* __restrict__ bias,
    const _Float16* __restrict__ Wt2, const float* __restrict__ bias2,
    const float* __restrict__ aux,
    float* __restrict__ outF, _Float16* __restrict__ outH,
    int Nout, int M) {
    const int lane = threadIdx.x & 31;
    const int wave = threadIdx.x >> 5;
    const int row0 = blockIdx.x * 128 + wave * 16;
    const int colB = blockIdx.y * 128;
    if (row0 >= M) return;

    v8f acc[8]  = {};
    v8f acc2[8] = {};

    for (int k0 = 0; k0 < Kin; k0 += 32) {
        v16h af = load_a_frag(A, Kin, row0, k0, lane);
#pragma unroll
        for (int nt = 0; nt < 8; ++nt) {
            v16h bf = load_b_frag(Wt, Kin, colB + nt * 16, k0, lane);
            acc[nt] = __builtin_amdgcn_wmma_f32_16x16x32_f16(
                false, af, false, bf, (short)0, acc[nt], false, false);
            if constexpr (MODE == 2) {
                v16h bf2 = load_b_frag(Wt2, Kin, colB + nt * 16, k0, lane);
                acc2[nt] = __builtin_amdgcn_wmma_f32_16x16x32_f16(
                    false, af, false, bf2, (short)0, acc2[nt], false, false);
            }
        }
    }

    const int n  = lane & 15;
    const int mb = (lane >> 4) * 8;
#pragma unroll
    for (int nt = 0; nt < 8; ++nt) {
        const int col = colB + nt * 16 + n;
        const float bs  = bias[col];
        const float bs2 = (MODE == 2) ? bias2[col] : 0.0f;
#pragma unroll
        for (int r = 0; r < 8; ++r) {
            const size_t idx = (size_t)(row0 + mb + r) * Nout + col;
            float y = acc[nt][r] + bs;
            if constexpr (MODE == 0) {
                outF[idx] = y;
            } else if constexpr (MODE == 1) {
                float ge = 0.5f * y * (1.0f + erff(y * 0.70710678118654752f));
                outH[idx] = (_Float16)ge;
            } else if constexpr (MODE == 2) {
                float g = acc2[nt][r] + bs2;
                outF[idx] = y * (1.0f / (1.0f + __expf(-g)));
            } else if constexpr (MODE == 3) {
                float s = 1.0f / (1.0f + __expf(-y));
                outF[idx] += s * aux[idx];
            } else {
                outF[idx] += y;
            }
        }
    }
}

// ---------------------------------------------------------------------------
// Weight pack: W[fin,fout] f32 -> Wt[fout,fin] f16 (N-major, B-frag friendly)
// ---------------------------------------------------------------------------
__global__ void pack_w(const float* __restrict__ W, _Float16* __restrict__ out,
                       int fin, int fout) {
    int idx = blockIdx.x * 256 + threadIdx.x;
    if (idx >= fin * fout) return;
    int o = idx / fin, i = idx - o * fin;
    out[idx] = (_Float16)W[(size_t)i * fout + o];
}

// ---------------------------------------------------------------------------
// LayerNorm over C=128, one wave per row, f16 output (feeds WMMA A operand)
// ---------------------------------------------------------------------------
__device__ __forceinline__ float wave_sum(float s) {
#pragma unroll
    for (int m = 16; m >= 1; m >>= 1) s += __shfl_xor(s, m, 32);
    return s;
}

__global__ __launch_bounds__(256) void ln128(const float* __restrict__ in,
                                             const float* __restrict__ g,
                                             const float* __restrict__ b,
                                             _Float16* __restrict__ outh, int M) {
    const int lane = threadIdx.x & 31;
    const int wave = threadIdx.x >> 5;
    const int row  = blockIdx.x * 8 + wave;
    if (row >= M) return;
    const int c = lane * 4;
    const float* p = in + (size_t)row * CDIM + c;
    float x0 = p[0], x1 = p[1], x2 = p[2], x3 = p[3];
    float mean = wave_sum(x0 + x1 + x2 + x3) * (1.0f / 128.0f);
    float d0 = x0 - mean, d1 = x1 - mean, d2 = x2 - mean, d3 = x3 - mean;
    float var = wave_sum(d0 * d0 + d1 * d1 + d2 * d2 + d3 * d3) * (1.0f / 128.0f);
    float inv = rsqrtf(var + 1e-5f);
    v4h o;
    o[0] = (_Float16)(d0 * inv * g[c + 0] + b[c + 0]);
    o[1] = (_Float16)(d1 * inv * g[c + 1] + b[c + 1]);
    o[2] = (_Float16)(d2 * inv * g[c + 2] + b[c + 2]);
    o[3] = (_Float16)(d3 * inv * g[c + 3] + b[c + 3]);
    *(v4h*)(outh + (size_t)row * CDIM + c) = o;
}

// ---------------------------------------------------------------------------
// KNN: one block (128 thr) per node; register top-12 + LDS tree merge
// ---------------------------------------------------------------------------
__global__ __launch_bounds__(128) void knn_kernel(const float* __restrict__ coords,
                                                  int* __restrict__ nbr,
                                                  float* __restrict__ kdist) {
    const int i   = blockIdx.x;
    const int tid = threadIdx.x;
    const float cx = coords[i * 3 + 0];
    const float cy = coords[i * 3 + 1];
    const float cz = coords[i * 3 + 2];

    float bd[KNN];
    int   bi[KNN];
#pragma unroll
    for (int t = 0; t < KNN; ++t) { bd[t] = 3.4e38f; bi[t] = 0x7fffffff; }

    for (int j = tid; j < NPTS; j += 128) {
        float dx = coords[j * 3 + 0] - cx;
        float dy = coords[j * 3 + 1] - cy;
        float dz = coords[j * 3 + 2] - cz;
        float d2 = dx * dx + dy * dy + dz * dz;
        if (d2 < bd[KNN - 1] || (d2 == bd[KNN - 1] && j < bi[KNN - 1])) {
            bd[KNN - 1] = d2; bi[KNN - 1] = j;
#pragma unroll
            for (int t = KNN - 1; t > 0; --t) {
                bool sw = (bd[t] < bd[t - 1]) ||
                          (bd[t] == bd[t - 1] && bi[t] < bi[t - 1]);
                if (sw) {
                    float td = bd[t]; bd[t] = bd[t - 1]; bd[t - 1] = td;
                    int   ti = bi[t]; bi[t] = bi[t - 1]; bi[t - 1] = ti;
                }
            }
        }
    }

    __shared__ float sd[128][KNN];
    __shared__ int   si[128][KNN];
#pragma unroll
    for (int t = 0; t < KNN; ++t) { sd[tid][t] = bd[t]; si[tid][t] = bi[t]; }

    for (int s = 64; s >= 1; s >>= 1) {
        __syncthreads();
        if (tid < s) {
            float od[KNN]; int oi[KNN];
            int pa = 0, pb = 0;
#pragma unroll
            for (int t = 0; t < KNN; ++t) {
                float da = (pa < KNN) ? sd[tid][pa]     : 3.4e38f;
                int   ia = (pa < KNN) ? si[tid][pa]     : 0x7fffffff;
                float db = (pb < KNN) ? sd[tid + s][pb] : 3.4e38f;
                int   ib = (pb < KNN) ? si[tid + s][pb] : 0x7fffffff;
                bool takeA = (da < db) || (da == db && ia <= ib);
                od[t] = takeA ? da : db;
                oi[t] = takeA ? ia : ib;
                if (takeA) ++pa; else ++pb;
            }
#pragma unroll
            for (int t = 0; t < KNN; ++t) { sd[tid][t] = od[t]; si[tid][t] = oi[t]; }
        }
    }
    __syncthreads();
    if (tid < KNN) {
        nbr[i * KNN + tid]   = si[0][tid];
        kdist[i * KNN + tid] = sqrtf(fmaxf(sd[0][tid], 0.0f));
    }
}

// RBF embedding: v[e,c] = thresholded gaussian bank
__global__ void embed_kernel(const float* __restrict__ kdist, float* __restrict__ v) {
    int idx = blockIdx.x * 256 + threadIdx.x;
    if (idx >= MEDGE * CDIM) return;
    int c = idx & (CDIM - 1);
    int e = idx >> 7;
    float d   = kdist[e];
    float sig = 1.0f + 4.0f * (float)c * (1.0f / 127.0f);
    float g   = __expf(-(d * d) / (2.0f * sig * sig));
    v[idx] = (g > 0.1f) ? g : 0.0f;
}

// Reciprocal-edge flat index: tIdx[e(i,k)] = flat index of edge (j,i) or -1
__global__ void tidx_kernel(const int* __restrict__ nbr, int* __restrict__ tIdx) {
    int e = blockIdx.x * 256 + threadIdx.x;
    if (e >= MEDGE) return;
    int i = e / KNN;
    int j = nbr[e];
    int r = -1;
#pragma unroll
    for (int t = 0; t < KNN; ++t)
        if (r < 0 && nbr[j * KNN + t] == i) r = j * KNN + t;
    tIdx[e] = r;
}

// transpose_vals gather
__global__ void tgather_kernel(const float* __restrict__ in,
                               const int* __restrict__ tIdx,
                               float* __restrict__ out) {
    int idx = blockIdx.x * 256 + threadIdx.x;
    if (idx >= MEDGE * CDIM) return;
    int e = idx >> 7, c = idx & (CDIM - 1);
    int t = tIdx[e];
    out[idx] = (t >= 0) ? in[(size_t)t * CDIM + c] : 0.0f;
}

// k[n,m,c] = sum_t a[n,t,c] * b[nbr[n,m],t,c]   (one block of 128 per node)
__global__ __launch_bounds__(128) void einsum_kernel(const float* __restrict__ a,
                                                     const float* __restrict__ b,
                                                     const int* __restrict__ nbr,
                                                     float* __restrict__ kout) {
    const int n = blockIdx.x;
    const int c = threadIdx.x;
    float ar[KNN];
#pragma unroll
    for (int t = 0; t < KNN; ++t)
        ar[t] = a[((size_t)n * KNN + t) * CDIM + c];
    __shared__ int nb[KNN];
    if (c < KNN) nb[c] = nbr[n * KNN + c];
    __syncthreads();
    for (int m = 0; m < KNN; ++m) {
        const float* bp = b + (size_t)nb[m] * KNN * CDIM + c;
        float s = 0.0f;
#pragma unroll
        for (int t = 0; t < KNN; ++t) s += ar[t] * bp[(size_t)t * CDIM];
        kout[((size_t)n * KNN + m) * CDIM + c] = s;
    }
}

// decode: e[edge] = dot(v[edge]+v[tIdx[edge]], decW) + decb  (wave per edge)
__global__ __launch_bounds__(256) void decode_kernel(const float* __restrict__ v,
                                                     const int* __restrict__ tIdx,
                                                     const float* __restrict__ decW,
                                                     const float* __restrict__ decb,
                                                     float* __restrict__ e) {
    const int lane = threadIdx.x & 31;
    const int wave = threadIdx.x >> 5;
    const int edge = blockIdx.x * 8 + wave;
    if (edge >= MEDGE) return;
    const int t = tIdx[edge];
    const float* pv = v + (size_t)edge * CDIM;
    const float* pt = v + (size_t)(t >= 0 ? t : 0) * CDIM;
    float s = 0.0f;
#pragma unroll
    for (int q = 0; q < 4; ++q) {
        int c = lane + q * 32;
        float x = pv[c] + ((t >= 0) ? pt[c] : 0.0f);
        s += x * decW[c];
    }
    s = wave_sum(s);
    if (lane == 0) e[edge] = s + decb[0];
}

__global__ void zero_kernel(float* __restrict__ p, size_t n) {
    size_t i = (size_t)blockIdx.x * 256 + threadIdx.x;
    if (i < n) p[i] = 0.0f;
}

__global__ void scatter_kernel(const float* __restrict__ e,
                               const int* __restrict__ nbr,
                               float* __restrict__ dense) {
    int edge = blockIdx.x * 256 + threadIdx.x;
    if (edge >= MEDGE) return;
    int i = edge / KNN;
    int j = nbr[edge];
    atomicAdd(dense + (size_t)i * NPTS + j, e[edge]);
}

// ---------------------------------------------------------------------------
// Host orchestration
// ---------------------------------------------------------------------------
extern "C" void kernel_launch(void* const* d_in, const int* in_sizes, int n_in,
                              void* d_out, int out_size, void* d_ws, size_t ws_size,
                              hipStream_t stream) {
    (void)in_sizes; (void)n_in; (void)out_size; (void)ws_size;
    const int M = MEDGE;

    // Inputs in setup_inputs() dict order
    const float* coords   = (const float*)d_in[0];
    const float* tri_ln_g = (const float*)d_in[1];
    const float* tri_ln_b = (const float*)d_in[2];
    const float* tri_Wa   = (const float*)d_in[3];
    const float* tri_ba   = (const float*)d_in[4];
    const float* tri_Wga  = (const float*)d_in[5];
    const float* tri_bga  = (const float*)d_in[6];
    const float* tri_Wb   = (const float*)d_in[7];
    const float* tri_bb   = (const float*)d_in[8];
    const float* tri_Wgb  = (const float*)d_in[9];
    const float* tri_bgb  = (const float*)d_in[10];
    const float* tri_Wo   = (const float*)d_in[11];
    const float* tri_bo   = (const float*)d_in[12];
    const float* tri_Wgo  = (const float*)d_in[13];
    const float* tri_bgo  = (const float*)d_in[14];
    const float* tri_lnog = (const float*)d_in[15];
    const float* tri_lnob = (const float*)d_in[16];
    const float* ffn_ln_g = (const float*)d_in[17];
    const float* ffn_ln_b = (const float*)d_in[18];
    const float* ffn_W1   = (const float*)d_in[19];
    const float* ffn_b1   = (const float*)d_in[20];
    const float* ffn_W2   = (const float*)d_in[21];
    const float* ffn_b2   = (const float*)d_in[22];
    const float* dec_W    = (const float*)d_in[23];
    const float* dec_b    = (const float*)d_in[24];

    // Workspace carve-up (256B aligned; B3,B4 contiguous so h(f16) can span them)
    char* base = (char*)d_ws;
    size_t off = 0;
    auto carve = [&](size_t bytes) -> char* {
        char* p = base + off;
        off = (off + bytes + 255) & ~(size_t)255;
        return p;
    };
    const size_t fbuf = (size_t)M * CDIM * sizeof(float);
    float*     v    = (float*)carve(fbuf);
    float*     B2   = (float*)carve(fbuf);
    float*     B3   = (float*)carve(fbuf);   // hbuf (f16, M*512) aliases B3..B4
    float*     B4   = (float*)carve(fbuf);
    _Float16*  xh   = (_Float16*)carve((size_t)M * CDIM * 2);
    _Float16*  kh   = (_Float16*)carve((size_t)M * CDIM * 2);
    int*       nbr  = (int*)carve((size_t)M * 4);
    int*       tIdx = (int*)carve((size_t)M * 4);
    float*     kdst = (float*)carve((size_t)M * 4);
    float*     ebuf = (float*)carve((size_t)M * 4);
    _Float16*  pTri = (_Float16*)carve((size_t)LLAYERS * 2 * 6 * CDIM * CDIM * 2);
    _Float16*  pW1  = (_Float16*)carve((size_t)LLAYERS * CDIM * FFDIM * 2);
    _Float16*  pW2  = (_Float16*)carve((size_t)LLAYERS * FFDIM * CDIM * 2);
    _Float16*  hbuf = (_Float16*)B3;         // M*512 f16 == exactly B3+B4

    // ---- pack all weights to f16, N-major ----
    const float* triW[6] = { tri_Wa, tri_Wga, tri_Wb, tri_Wgb, tri_Wo, tri_Wgo };
    for (int lr = 0; lr < LLAYERS * 2; ++lr) {
        for (int w = 0; w < 6; ++w) {
            pack_w<<<dim3((CDIM * CDIM) / 256), 256, 0, stream>>>(
                triW[w] + (size_t)lr * CDIM * CDIM,
                pTri + ((size_t)lr * 6 + w) * CDIM * CDIM, CDIM, CDIM);
        }
    }
    for (int l = 0; l < LLAYERS; ++l) {
        pack_w<<<dim3((CDIM * FFDIM) / 256), 256, 0, stream>>>(
            ffn_W1 + (size_t)l * CDIM * FFDIM, pW1 + (size_t)l * CDIM * FFDIM,
            CDIM, FFDIM);
        pack_w<<<dim3((FFDIM * CDIM) / 256), 256, 0, stream>>>(
            ffn_W2 + (size_t)l * FFDIM * CDIM, pW2 + (size_t)l * FFDIM * CDIM,
            FFDIM, CDIM);
    }

    // ---- KNN embedding ----
    knn_kernel<<<dim3(NPTS), 128, 0, stream>>>(coords, nbr, kdst);
    tidx_kernel<<<dim3((MEDGE + 255) / 256), 256, 0, stream>>>(nbr, tIdx);
    embed_kernel<<<dim3((MEDGE * CDIM) / 256), 256, 0, stream>>>(kdst, v);

    const dim3 gGemm1(M / 128, 1), gGemm4(M / 128, FFDIM / 128);
    const dim3 gElem((MEDGE * CDIM) / 256), gRow(M / 8);

    // ---- layers ----
    for (int l = 0; l < LLAYERS; ++l) {
        for (int rc = 0; rc < 2; ++rc) {
            const int lr = l * 2 + rc;
            const _Float16* pWa  = pTri + ((size_t)lr * 6 + 0) * CDIM * CDIM;
            const _Float16* pWga = pTri + ((size_t)lr * 6 + 1) * CDIM * CDIM;
            const _Float16* pWb  = pTri + ((size_t)lr * 6 + 2) * CDIM * CDIM;
            const _Float16* pWgb = pTri + ((size_t)lr * 6 + 3) * CDIM * CDIM;
            const _Float16* pWo  = pTri + ((size_t)lr * 6 + 4) * CDIM * CDIM;
            const _Float16* pWgo = pTri + ((size_t)lr * 6 + 5) * CDIM * CDIM;

            // x = LN(v) (f16)
            ln128<<<gRow, 256, 0, stream>>>(v, tri_ln_g + (size_t)lr * CDIM,
                                            tri_ln_b + (size_t)lr * CDIM, xh, M);
            // a = sigmoid(x@Wga+bga)*(x@Wa+ba) -> B2 ; b likewise -> B3
            gemm128<2><<<gGemm1, 256, 0, stream>>>(
                xh, CDIM, pWa, tri_ba + (size_t)lr * CDIM,
                pWga, tri_bga + (size_t)lr * CDIM, nullptr, B2, nullptr, CDIM, M);
            gemm128<2><<<gGemm1, 256, 0, stream>>>(
                xh, CDIM, pWb, tri_bb + (size_t)lr * CDIM,
                pWgb, tri_bgb + (size_t)lr * CDIM, nullptr, B3, nullptr, CDIM, M);

            const float *pa, *pb;
            float* pk;
            if (rc == 1) {  // incoming: transpose a, b to reciprocal-edge slots
                tgather_kernel<<<gElem, 256, 0, stream>>>(B2, tIdx, B4);
                tgather_kernel<<<gElem, 256, 0, stream>>>(B3, tIdx, B2);
                pa = B4; pb = B2; pk = B3;
            } else {
                pa = B2; pb = B3; pk = B4;
            }
            // triangle einsum over KNN slots
            einsum_kernel<<<dim3(NPTS), 128, 0, stream>>>(pa, pb, nbr, pk);
            // o = LN(k) @ Wo + bo  -> B2 (free in both branches)
            ln128<<<gRow, 256, 0, stream>>>(pk, tri_lnog + (size_t)lr * CDIM,
                                            tri_lnob + (size_t)lr * CDIM, kh, M);
            gemm128<0><<<gGemm1, 256, 0, stream>>>(
                kh, CDIM, pWo, tri_bo + (size_t)lr * CDIM,
                nullptr, nullptr, nullptr, B2, nullptr, CDIM, M);
            // v += sigmoid(x@Wgo+bgo) * o
            gemm128<3><<<gGemm1, 256, 0, stream>>>(
                xh, CDIM, pWgo, tri_bgo + (size_t)lr * CDIM,
                nullptr, nullptr, B2, v, nullptr, CDIM, M);
        }
        // FFN: v += gelu(LN(v)@W1+b1)@W2 + b2
        ln128<<<gRow, 256, 0, stream>>>(v, ffn_ln_g + (size_t)l * CDIM,
                                        ffn_ln_b + (size_t)l * CDIM, xh, M);
        gemm128<1><<<gGemm4, 256, 0, stream>>>(
            xh, CDIM, pW1 + (size_t)l * CDIM * FFDIM, ffn_b1 + (size_t)l * FFDIM,
            nullptr, nullptr, nullptr, nullptr, hbuf, FFDIM, M);
        gemm128<4><<<gGemm1, 256, 0, stream>>>(
            hbuf, FFDIM, pW2 + (size_t)l * FFDIM * CDIM, ffn_b2 + (size_t)l * CDIM,
            nullptr, nullptr, nullptr, v, nullptr, CDIM, M);
    }

    // ---- decode + dense scatter ----
    decode_kernel<<<gRow, 256, 0, stream>>>(v, tIdx, dec_W, dec_b, ebuf);
    zero_kernel<<<dim3((unsigned)(((size_t)NPTS * NPTS + 255) / 256)), 256, 0, stream>>>(
        (float*)d_out, (size_t)NPTS * NPTS);
    scatter_kernel<<<dim3((MEDGE + 255) / 256), 256, 0, stream>>>(
        ebuf, nbr, (float*)d_out);
}
